// CosBlock_33861522162476
// MI455X (gfx1250) — matrix operand order repaired and checked by reading, not compiled, not run on hardware
//
#include <hip/hip_runtime.h>
#include <hip/hip_bf16.h>
#include <math.h>

typedef __attribute__((ext_vector_type(16))) _Float16 v16h;
typedef __attribute__((ext_vector_type(8)))  float    v8f;
typedef __attribute__((ext_vector_type(4)))  unsigned int v4u;

// Problem constants (match reference setup_inputs)
constexpr int Bq  = 2;
constexpr int Lq  = 2048;
constexpr int Dm  = 1024;
constexpr int Hh  = 16;
constexpr int Ee  = 64;      // Dm / Hh
constexpr int SK  = 40;      // sample_k = 5*ceil(log(2048))
constexpr int Uu  = 40;      // u = min(sample_k, L)
constexpr int DFF = 16;

union Frag {
    v16h v;
    v4u  q[2];
    unsigned int u[8];
};

// ---------------- utility reductions (256-thread blocks) ----------------
__device__ inline float blk_reduce_sum(float val, float* red) {
    red[threadIdx.x] = val; __syncthreads();
    for (int s = 128; s > 0; s >>= 1) {
        if (threadIdx.x < s) red[threadIdx.x] += red[threadIdx.x + s];
        __syncthreads();
    }
    float r = red[0]; __syncthreads();
    return r;
}
__device__ inline float blk_reduce_max(float val, float* red) {
    red[threadIdx.x] = val; __syncthreads();
    for (int s = 128; s > 0; s >>= 1) {
        if (threadIdx.x < s) red[threadIdx.x] = fmaxf(red[threadIdx.x], red[threadIdx.x + s]);
        __syncthreads();
    }
    float r = red[0]; __syncthreads();
    return r;
}

// ---------------- fp32 -> fp16 conversion ----------------
__global__ __launch_bounds__(256) void cvt_f16_kernel(const float* __restrict__ src,
                                                      _Float16* __restrict__ dst, int n) {
    int i = blockIdx.x * 256 + threadIdx.x;
    if (i < n) dst[i] = (_Float16)src[i];
}

// ---------------- LayerNorm0: x -> xn (f32) and xn_h (f16) ----------------
__global__ __launch_bounds__(256) void ln0_kernel(const float* __restrict__ x,
                                                  const float* __restrict__ g,
                                                  const float* __restrict__ bb,
                                                  float* __restrict__ xn,
                                                  _Float16* __restrict__ xnh) {
    __shared__ float red[256];
    int row = blockIdx.x;                 // B*L rows
    const float* xr = x + (size_t)row * Dm;
    float vals[4];
    float s = 0.f, s2 = 0.f;
    #pragma unroll
    for (int i = 0; i < 4; i++) {
        int c = threadIdx.x + i * 256;
        float t = xr[c];
        vals[i] = t; s += t; s2 += t * t;
    }
    float mean = blk_reduce_sum(s, red) * (1.0f / Dm);
    float ms2  = blk_reduce_sum(s2, red) * (1.0f / Dm);
    float inv  = rsqrtf(ms2 - mean * mean + 1e-5f);
    #pragma unroll
    for (int i = 0; i < 4; i++) {
        int c = threadIdx.x + i * 256;
        float y = (vals[i] - mean) * inv * g[c] + bb[c];
        xn[(size_t)row * Dm + c]  = y;
        xnh[(size_t)row * Dm + c] = (_Float16)y;
    }
}

// ---------------- WMMA GEMM: out[m,n] = sum_k A[m,k]*W[n,k] + bias[n] ----------------
// A: (M,K) f16 row-major.  W: (N,K) f16 row-major (i.e. computing A @ W^T).
// Block = 256 threads = 8 waves; wave computes a 16x64 output tile (4 accumulators).
__global__ __launch_bounds__(256) void gemm_wmma_kernel(const _Float16* __restrict__ A,
                                                        const _Float16* __restrict__ W,
                                                        const float* __restrict__ bias,
                                                        float* __restrict__ out,
                                                        int Mdim, int Ndim, int Kdim) {
    int wave = threadIdx.x >> 5;
    int lane = threadIdx.x & 31;
    int half = lane >> 4;                 // 0: lanes 0-15, 1: lanes 16-31
    int loc  = lane & 15;                 // row (A) / col (B) within 16-tile
    int m0 = blockIdx.y * 128 + wave * 16;
    int n0 = blockIdx.x * 64;

    v8f c0 = {}, c1 = {}, c2 = {}, c3 = {};

    const _Float16* arow = A + (size_t)(m0 + loc) * Kdim;
    const _Float16* wrow0 = W + (size_t)(n0 +  0 + loc) * Kdim;
    const _Float16* wrow1 = W + (size_t)(n0 + 16 + loc) * Kdim;
    const _Float16* wrow2 = W + (size_t)(n0 + 32 + loc) * Kdim;
    const _Float16* wrow3 = W + (size_t)(n0 + 48 + loc) * Kdim;

    for (int kk = 0; kk < Kdim; kk += 32) {
        // A fragment (16x32 f16): VGPR v -> K = kk + half*16 + 2v(+1)
        Frag a;
        a.q[0] = *(const v4u*)(arow + kk + half * 16);
        a.q[1] = *(const v4u*)(arow + kk + half * 16 + 8);

        // B fragment (32x16 f16): lane group half -> K base kk + half*16;
        // VGPR v -> K = base + 2v(+1); column n = loc.  B[k][n] = W[n][k].
        Frag b0, b1, b2, b3;
        int kb = kk + half * 16;
        b0.q[0] = *(const v4u*)(wrow0 + kb);     b0.q[1] = *(const v4u*)(wrow0 + kb + 8);
        b1.q[0] = *(const v4u*)(wrow1 + kb);     b1.q[1] = *(const v4u*)(wrow1 + kb + 8);
        b2.q[0] = *(const v4u*)(wrow2 + kb);     b2.q[1] = *(const v4u*)(wrow2 + kb + 8);
        b3.q[0] = *(const v4u*)(wrow3 + kb);     b3.q[1] = *(const v4u*)(wrow3 + kb + 8);

        c0 = __builtin_amdgcn_wmma_f32_16x16x32_f16(false, a.v, false, b0.v, (short)0, c0, false, false);
        c1 = __builtin_amdgcn_wmma_f32_16x16x32_f16(false, a.v, false, b1.v, (short)0, c1, false, false);
        c2 = __builtin_amdgcn_wmma_f32_16x16x32_f16(false, a.v, false, b2.v, (short)0, c2, false, false);
        c3 = __builtin_amdgcn_wmma_f32_16x16x32_f16(false, a.v, false, b3.v, (short)0, c3, false, false);
    }

    // C/D layout: VGPR r -> (M = m0 + r + half*8, N = n0 + j*16 + loc)
    v8f acc[4] = {c0, c1, c2, c3};
    #pragma unroll
    for (int j = 0; j < 4; j++) {
        int col = n0 + j * 16 + loc;
        float bv = bias[col];
        #pragma unroll
        for (int r = 0; r < 8; r++) {
            int rowi = m0 + r + half * 8;
            out[(size_t)rowi * Ndim + col] = acc[j][r] + bv;
        }
    }
}

// ---------------- v mean over L, and ctx broadcast-init ----------------
__global__ __launch_bounds__(256) void vmean_kernel(const float* __restrict__ v,
                                                    float* __restrict__ vmean) {
    int t = blockIdx.x * 256 + threadIdx.x;      // B*Dm threads
    if (t >= Bq * Dm) return;
    int b = t / Dm, c = t % Dm;
    float s = 0.f;
    for (int l = 0; l < Lq; l++) s += v[((size_t)(b * Lq + l)) * Dm + c];
    vmean[t] = s * (1.0f / Lq);
}

__global__ __launch_bounds__(256) void ctx_init_kernel(const float* __restrict__ vmean,
                                                       float* __restrict__ ctx) {
    size_t i = (size_t)blockIdx.x * 256 + threadIdx.x;   // B*L*D elems
    int c = (int)(i % Dm);
    int b = (int)(i / ((size_t)Lq * Dm));
    ctx[i] = vmean[b * Dm + c];
}

// ---------------- sampled sparsity measure M[b,h,l] ----------------
__global__ __launch_bounds__(256) void sample_kernel(const float* __restrict__ q,
                                                     const float* __restrict__ k,
                                                     const int* __restrict__ idx,
                                                     float* __restrict__ Mbuf) {
    int t = blockIdx.x * 256 + threadIdx.x;       // B*H*L threads
    if (t >= Bq * Hh * Lq) return;
    int h = t & (Hh - 1);
    int l = (t >> 4) & (Lq - 1);
    int b = t >> 15;
    const float* qv = q + ((size_t)(b * Lq + l)) * Dm + h * Ee;
    float mx = -3.0e38f, sm = 0.f;
    for (int s = 0; s < SK; s++) {
        int kl = idx[l * SK + s];
        const float* kv = k + ((size_t)(b * Lq + kl)) * Dm + h * Ee;
        float d = 0.f;
        #pragma unroll 8
        for (int e = 0; e < Ee; e++) d += qv[e] * kv[e];
        mx = fmaxf(mx, d);
        sm += d;
    }
    Mbuf[((size_t)(b * Hh + h)) * Lq + l] = mx - sm * (1.0f / Lq);
}

// ---------------- top-U selection per (b,h) ----------------
__global__ __launch_bounds__(256) void topk_kernel(const float* __restrict__ Mbuf,
                                                   int* __restrict__ topb) {
    int bh = blockIdx.x;
    __shared__ float sM[Lq];
    __shared__ float rv[256];
    __shared__ int   ri[256];
    for (int i = threadIdx.x; i < Lq; i += 256) sM[i] = Mbuf[(size_t)bh * Lq + i];
    __syncthreads();
    for (int u = 0; u < Uu; u++) {
        float bv = -3.0e38f; int bi = Lq;
        for (int i = threadIdx.x; i < Lq; i += 256) {
            float t = sM[i];
            if (t > bv || (t == bv && i < bi)) { bv = t; bi = i; }
        }
        rv[threadIdx.x] = bv; ri[threadIdx.x] = bi; __syncthreads();
        for (int s = 128; s > 0; s >>= 1) {
            if (threadIdx.x < s) {
                float ov = rv[threadIdx.x + s]; int oi = ri[threadIdx.x + s];
                if (ov > rv[threadIdx.x] || (ov == rv[threadIdx.x] && oi < ri[threadIdx.x])) {
                    rv[threadIdx.x] = ov; ri[threadIdx.x] = oi;
                }
            }
            __syncthreads();
        }
        if (threadIdx.x == 0) {
            topb[bh * Uu + u] = ri[0];
            sM[ri[0]] = -3.0e38f;
        }
        __syncthreads();
    }
}

// ---------------- attention for the U selected queries; scatter into ctx ----------------
__global__ __launch_bounds__(256) void attn_kernel(const float* __restrict__ q,
                                                   const float* __restrict__ k,
                                                   const float* __restrict__ v,
                                                   const int* __restrict__ topb,
                                                   float* __restrict__ ctx) {
    __shared__ float qs[Ee];
    __shared__ float aw[Lq];      // 8 KB attention weights
    __shared__ float red[256];
    __shared__ float part[4][Ee];

    int bh = blockIdx.x / Uu;
    int u  = blockIdx.x % Uu;
    int b = bh / Hh, h = bh % Hh;
    int lqi = topb[bh * Uu + u];

    if (threadIdx.x < Ee)
        qs[threadIdx.x] = q[((size_t)(b * Lq + lqi)) * Dm + h * Ee + threadIdx.x];
    __syncthreads();

    // scores
    float sc[8];
    float lmax = -3.0e38f;
    #pragma unroll
    for (int i = 0; i < 8; i++) {
        int l = threadIdx.x + i * 256;
        const float* kv = k + ((size_t)(b * Lq + l)) * Dm + h * Ee;
        float d = 0.f;
        #pragma unroll 8
        for (int e = 0; e < Ee; e++) d += qs[e] * kv[e];
        d *= 0.125f;                 // 1/sqrt(64)
        sc[i] = d;
        lmax = fmaxf(lmax, d);
    }
    float bmax = blk_reduce_max(lmax, red);
    float lsum = 0.f;
    #pragma unroll
    for (int i = 0; i < 8; i++) {
        float ex = __expf(sc[i] - bmax);
        aw[threadIdx.x + i * 256] = ex;
        lsum += ex;
    }
    float bsum = blk_reduce_sum(lsum, red);
    __syncthreads();

    // ctx_u[e] = sum_l aw[l] * v[b,l,h,e] / bsum
    int e   = threadIdx.x & 63;
    int grp = threadIdx.x >> 6;       // 4 groups
    float acc = 0.f;
    for (int l = grp; l < Lq; l += 4)
        acc += aw[l] * v[((size_t)(b * Lq + l)) * Dm + h * Ee + e];
    part[grp][e] = acc;
    __syncthreads();
    if (grp == 0) {
        float tot = part[0][e] + part[1][e] + part[2][e] + part[3][e];
        ctx[((size_t)(b * Lq + lqi)) * Dm + h * Ee + e] = tot / bsum;
    }
}

// ---------------- fused tail: h = LN1(xn + attn_out); y = FFN(h); out = LN2(h+y) ----------------
__global__ __launch_bounds__(256) void tail_kernel(const float* __restrict__ xn,
                                                   const float* __restrict__ ao,
                                                   const float* __restrict__ g1,
                                                   const float* __restrict__ b1,
                                                   const float* __restrict__ c1w,
                                                   const float* __restrict__ c1b,
                                                   const float* __restrict__ c2w,
                                                   const float* __restrict__ c2b,
                                                   const float* __restrict__ g2,
                                                   const float* __restrict__ b2,
                                                   float* __restrict__ out) {
    __shared__ float hrow[Dm];
    __shared__ float red[256];
    __shared__ float pp[DFF][17];
    __shared__ float y1[DFF];
    int row = blockIdx.x;

    float t4[4];
    float s = 0.f, s2 = 0.f;
    #pragma unroll
    for (int i = 0; i < 4; i++) {
        int c = threadIdx.x + i * 256;
        float tv = xn[(size_t)row * Dm + c] + ao[(size_t)row * Dm + c];
        t4[i] = tv; s += tv; s2 += tv * tv;
    }
    float mean = blk_reduce_sum(s, red) * (1.0f / Dm);
    float ms2  = blk_reduce_sum(s2, red) * (1.0f / Dm);
    float inv  = rsqrtf(ms2 - mean * mean + 1e-5f);
    #pragma unroll
    for (int i = 0; i < 4; i++) {
        int c = threadIdx.x + i * 256;
        hrow[c] = (t4[i] - mean) * inv * g1[c] + b1[c];
    }
    __syncthreads();

    // conv1 (DFF x Dm) + GELU(exact)
    int f  = threadIdx.x & 15;
    int ch = threadIdx.x >> 4;              // 16 chunks of 64
    float p = 0.f;
    int d0 = ch * 64;
    #pragma unroll 8
    for (int d = 0; d < 64; d++) p += hrow[d0 + d] * c1w[f * Dm + d0 + d];
    pp[f][ch] = p;
    __syncthreads();
    if (threadIdx.x < DFF) {
        float tot = 0.f;
        #pragma unroll
        for (int ci = 0; ci < 16; ci++) tot += pp[threadIdx.x][ci];
        tot += c1b[threadIdx.x];
        y1[threadIdx.x] = 0.5f * tot * (1.0f + erff(tot * 0.70710678118654752f));
    }
    __syncthreads();

    // conv2 (Dm x DFF) + residual + LN2
    float r4[4];
    s = 0.f; s2 = 0.f;
    #pragma unroll
    for (int i = 0; i < 4; i++) {
        int c = threadIdx.x + i * 256;
        float yv = c2b[c];
        #pragma unroll
        for (int ff = 0; ff < DFF; ff++) yv += y1[ff] * c2w[c * DFF + ff];
        float rv = hrow[c] + yv;
        r4[i] = rv; s += rv; s2 += rv * rv;
    }
    float mean2 = blk_reduce_sum(s, red) * (1.0f / Dm);
    float ms22  = blk_reduce_sum(s2, red) * (1.0f / Dm);
    float inv2  = rsqrtf(ms22 - mean2 * mean2 + 1e-5f);
    #pragma unroll
    for (int i = 0; i < 4; i++) {
        int c = threadIdx.x + i * 256;
        out[(size_t)row * Dm + c] = (r4[i] - mean2) * inv2 * g2[c] + b2[c];
    }
}

// ---------------- host-side orchestration ----------------
extern "C" void kernel_launch(void* const* d_in, const int* in_sizes, int n_in,
                              void* d_out, int out_size, void* d_ws, size_t ws_size,
                              hipStream_t stream) {
    (void)in_sizes; (void)n_in; (void)out_size; (void)ws_size;

    const float* x     = (const float*)d_in[0];
    const int*   idx   = (const int*)  d_in[1];
    const float* ln0g  = (const float*)d_in[2];
    const float* ln0b  = (const float*)d_in[3];
    const float* Wq    = (const float*)d_in[4];
    const float* bq    = (const float*)d_in[5];
    const float* Wk    = (const float*)d_in[6];
    const float* bk    = (const float*)d_in[7];
    const float* Wv    = (const float*)d_in[8];
    const float* bv    = (const float*)d_in[9];
    const float* Wo    = (const float*)d_in[10];
    const float* bo    = (const float*)d_in[11];
    const float* ln1g  = (const float*)d_in[12];
    const float* ln1b  = (const float*)d_in[13];
    const float* c1w   = (const float*)d_in[14];
    const float* c1b   = (const float*)d_in[15];
    const float* c2w   = (const float*)d_in[16];
    const float* c2b   = (const float*)d_in[17];
    const float* ln2g  = (const float*)d_in[18];
    const float* ln2b  = (const float*)d_in[19];
    float* out = (float*)d_out;

    const size_t BLD = (size_t)Bq * Lq * Dm;       // 4,194,304
    const size_t DD  = (size_t)Dm * Dm;            // 1,048,576

    char* ws = (char*)d_ws;
    size_t off = 0;
    auto alloc = [&](size_t bytes) -> void* {
        off = (off + 255) & ~(size_t)255;
        void* p = ws + off;
        off += bytes;
        return p;
    };

    float*    xn   = (float*)   alloc(BLD * 4);
    _Float16* xnh  = (_Float16*)alloc(BLD * 2);
    float*    qb   = (float*)   alloc(BLD * 4);
    float*    kb   = (float*)   alloc(BLD * 4);
    float*    vb   = (float*)   alloc(BLD * 4);
    _Float16* Wqh  = (_Float16*)alloc(DD * 2);
    _Float16* Wkh  = (_Float16*)alloc(DD * 2);
    _Float16* Wvh  = (_Float16*)alloc(DD * 2);
    _Float16* Woh  = (_Float16*)alloc(DD * 2);
    float*    Mbuf = (float*)   alloc((size_t)Bq * Hh * Lq * 4);
    int*      topb = (int*)     alloc((size_t)Bq * Hh * Uu * 4);
    float*    vmn  = (float*)   alloc((size_t)Bq * Dm * 4);
    float*    ctx  = (float*)   alloc(BLD * 4);
    _Float16* ctxh = (_Float16*)alloc(BLD * 2);
    float*    ao   = (float*)   alloc(BLD * 4);

    const int Mrows = Bq * Lq;                     // 4096
    const dim3 blk(256);

    // 1) weights -> f16
    {
        int n = (int)DD;
        int g = (n + 255) / 256;
        cvt_f16_kernel<<<g, blk, 0, stream>>>(Wq, Wqh, n);
        cvt_f16_kernel<<<g, blk, 0, stream>>>(Wk, Wkh, n);
        cvt_f16_kernel<<<g, blk, 0, stream>>>(Wv, Wvh, n);
        cvt_f16_kernel<<<g, blk, 0, stream>>>(Wo, Woh, n);
    }

    // 2) LN0
    ln0_kernel<<<Mrows, blk, 0, stream>>>(x, ln0g, ln0b, xn, xnh);

    // 3) Q,K,V projections (WMMA)
    {
        dim3 grid(Dm / 64, Mrows / 128);
        gemm_wmma_kernel<<<grid, blk, 0, stream>>>(xnh, Wqh, bq, qb, Mrows, Dm, Dm);
        gemm_wmma_kernel<<<grid, blk, 0, stream>>>(xnh, Wkh, bk, kb, Mrows, Dm, Dm);
        gemm_wmma_kernel<<<grid, blk, 0, stream>>>(xnh, Wvh, bv, vb, Mrows, Dm, Dm);
    }

    // 4) v mean + ctx broadcast init
    vmean_kernel<<<(Bq * Dm + 255) / 256, blk, 0, stream>>>(vb, vmn);
    ctx_init_kernel<<<(int)(BLD / 256), blk, 0, stream>>>(vmn, ctx);

    // 5) sampled sparsity measure
    sample_kernel<<<(Bq * Hh * Lq + 255) / 256, blk, 0, stream>>>(qb, kb, idx, Mbuf);

    // 6) top-U per (b,h)
    topk_kernel<<<Bq * Hh, blk, 0, stream>>>(Mbuf, topb);

    // 7) attention for selected queries, scatter into ctx
    attn_kernel<<<Bq * Hh * Uu, blk, 0, stream>>>(qb, kb, vb, topb, ctx);

    // 8) output projection (WMMA)
    cvt_f16_kernel<<<(int)(BLD / 256), blk, 0, stream>>>(ctx, ctxh, (int)BLD);
    {
        dim3 grid(Dm / 64, Mrows / 128);
        gemm_wmma_kernel<<<grid, blk, 0, stream>>>(ctxh, Woh, bo, ao, Mrows, Dm, Dm);
    }

    // 9) fused LN1 + FFN + LN2
    tail_kernel<<<Mrows, blk, 0, stream>>>(xn, ao, ln1g, ln1b, c1w, c1b, c2w, c2b,
                                           ln2g, ln2b, out);
}
